// gcn_lstm_41369124995480
// MI455X (gfx1250) — matrix operand order repaired and checked by reading, not compiled, not run on hardware
//
#include <hip/hip_runtime.h>
#include <math.h>

// ---------------------------------------------------------------------------
// GCN(2 fused prop layers) -> LSTM -> 5 FC heads for gfx1250 (CDNA5, wave32)
// Dominant 176 GFLOP runs on v_wmma_f32_16x16x32_bf16; per-bt diag(seq)
// scaling applied with packed v_pk_mul_bf16 so the W B-fragment is reused
// across 16 batch-time steps per workgroup (16x less L2 traffic).
// ---------------------------------------------------------------------------

typedef __attribute__((ext_vector_type(16))) __bf16    v16bf;
typedef __attribute__((ext_vector_type(8)))  float     v8f;
typedef __attribute__((ext_vector_type(8)))  unsigned  v8u;

#define N_NODES 1329
#define F_FEAT  130
#define B_BATCH 16
#define T_STEPS 24
#define BT      (B_BATCH * T_STEPS)            // 384
#define H_HID   15
#define G4H     (4 * H_HID)                    // 60
#define K_HEADS 5

#define KT      32                             // WMMA K depth (bf16)
#define MTILES  ((N_NODES + 15) / 16)          // 84
#define NFULL   ((N_NODES / KT) * KT)          // 1312 (tail = 17)
#define FPAD    144                            // F padded to 9 tiles of 16
#define NPAD    (((N_NODES + KT - 1) / KT) * KT)   // 1344
#define BTG     16                             // bt per stage-1 workgroup

__device__ __forceinline__ unsigned short f32_to_bf16_bits(float f) {
  unsigned u = __builtin_bit_cast(unsigned, f);
  u += 0x7fffu + ((u >> 16) & 1u);             // round-to-nearest-even
  return (unsigned short)(u >> 16);
}
__device__ __forceinline__ __bf16 bits_to_bf16(unsigned short h) {
  return __builtin_bit_cast(__bf16, h);
}
__device__ __forceinline__ __bf16 f2bf(float f) { return bits_to_bf16(f32_to_bf16_bits(f)); }
__device__ __forceinline__ float sigf(float x) { return 1.0f / (1.0f + __expf(-x)); }

// K order this code uses for 16-bit A and B fragments: element e of lane-half h
// holds K = kidxA(e,h) = 16h + e (contiguous 16-element run).
__device__ __forceinline__ int kidxA(int e, int half) {
  return e + (half << 4);
}

// ---------------------------------------------------------------------------
// Kernel 0: rowsumA[m] = sum_n A_hat[m,n]  (folds the GCN bias through A_hat)
// ---------------------------------------------------------------------------
__global__ void rowsum_kernel(const float* __restrict__ A, float* __restrict__ rs) {
  int m = blockIdx.x * blockDim.x + threadIdx.x;
  if (m >= N_NODES) return;
  const float* row = A + (size_t)m * N_NODES;
  float s = 0.0f;
  for (int n = 0; n < N_NODES; ++n) s += row[n];
  rs[m] = s;
}

// ---------------------------------------------------------------------------
// Stage-1 helpers
// ---------------------------------------------------------------------------
template <bool FULL>
__device__ __forceinline__ void s1_stage(
    int tid, int n0, int mBase, int btBase,
    const float* __restrict__ A, const float* __restrict__ W,
    const float* __restrict__ seq,
    unsigned short* ldsA, unsigned* ldsWT, unsigned short* ldsS)
{
  // raw A tile 16x32, natural [m][k] order
  for (int idx = tid; idx < 16 * KT; idx += 288) {
    int m = idx >> 5, k = idx & 31;
    int gm = mBase + m, gn = n0 + k;
    float v = 0.0f;
    if (FULL || (gm < N_NODES && gn < N_NODES))
      v = A[(size_t)gm * N_NODES + gn];
    ldsA[idx] = f32_to_bf16_bits(v);
  }
  // seq scale slab BTG x 32, natural [bt][k] order
  for (int idx = tid; idx < BTG * KT; idx += 288) {
    int b = idx >> 5, k = idx & 31;
    int gn = n0 + k;
    float v = 0.0f;
    if (FULL || gn < N_NODES)
      v = seq[(size_t)(btBase + b) * N_NODES + gn];
    ldsS[idx] = f32_to_bf16_bits(v);
  }
  // W slab, TRANSPOSED [f][k] so B-fragments are contiguous ds_load_b128s.
  {
    const int f  = tid % FPAD;                 // 0..143
    const int kh = tid / FPAD;                 // 0..1 -> k range [16kh,16kh+16)
    const bool fv = (f < F_FEAT);
#pragma unroll
    for (int j = 0; j < 8; ++j) {
      int k = kh * 16 + 2 * j;
      float v0 = 0.0f, v1 = 0.0f;
      if (fv) {
        int g0 = n0 + k;
        if (FULL || g0 < N_NODES)     v0 = W[(size_t)g0 * F_FEAT + f];
        if (FULL || g0 + 1 < N_NODES) v1 = W[(size_t)(g0 + 1) * F_FEAT + f];
      }
      ldsWT[(f * KT + k) >> 1] =
          (unsigned)f32_to_bf16_bits(v0) | ((unsigned)f32_to_bf16_bits(v1) << 16);
    }
  }
}

__device__ __forceinline__ void s1_compute(
    int wave, int half, int r,
    const unsigned short* ldsA, const unsigned* ldsWT, const unsigned short* ldsS,
    v8f (&acc)[BTG])
{
  // raw A fragment (2x ds_load_b128) -- shared by all bt
  v16bf araw;
#pragma unroll
  for (int e = 0; e < 16; ++e)
    araw[e] = bits_to_bf16(ldsA[r * KT + kidxA(e, half)]);
  const v8u au = __builtin_bit_cast(v8u, araw);

  // B fragment: 32 contiguous bytes of transposed W row -- reused for all bt
  const unsigned* wrow = ldsWT + ((wave * 16 + r) * KT >> 1) + half * 8;
  v8u bu;
#pragma unroll
  for (int j = 0; j < 8; ++j) bu[j] = wrow[j];
  const v16bf bfrag = __builtin_bit_cast(v16bf, bu);

#pragma unroll
  for (int bt = 0; bt < BTG; ++bt) {
    v16bf spk;
#pragma unroll
    for (int e = 0; e < 16; ++e)
      spk[e] = bits_to_bf16(ldsS[bt * KT + kidxA(e, half)]);
    const v8u su = __builtin_bit_cast(v8u, spk);
    v8u pu;
#pragma unroll
    for (int j = 0; j < 8; ++j) {                   // A' = A (.) diag(seq_bt)
      unsigned x = au[j], y = su[j], z;
      asm("v_pk_mul_bf16 %0, %1, %2" : "=v"(z) : "v"(x), "v"(y));
      pu[j] = z;
    }
    const v16bf afrag = __builtin_bit_cast(v16bf, pu);
    acc[bt] = __builtin_amdgcn_wmma_f32_16x16x32_bf16(false, afrag, false, bfrag,
                                                      (short)0, acc[bt], false, false);
  }
}

// ---------------------------------------------------------------------------
// Kernel 1 (dominant): block = (m-tile, group of 16 bt). 288 threads = 9
// waves, wave w owns f-tile w. Per bt:
//   AX[m,f] = sum_n A[m,n]*seq[bt,n]*W[n,f] + rowsumA[m]*bias[f]
//   Y1[bt,m] = sum_f relu(AX[m,f]) * weight2[f] + bias2
// ---------------------------------------------------------------------------
__global__ __launch_bounds__(288) void gcn_stage1_kernel(
    const float* __restrict__ seq,       // [BT, N]
    const float* __restrict__ A,         // [N, N]
    const float* __restrict__ W,         // [N, F]
    const float* __restrict__ bias,      // [F]
    const float* __restrict__ w2,        // [F]
    const float* __restrict__ b2,        // [1]
    const float* __restrict__ rowsumA,   // [N]
    float* __restrict__ Y1)              // [BT, N]
{
  __shared__ unsigned short ldsA[16 * KT];
  __shared__ unsigned       ldsWT[(FPAD * KT) / 2];
  __shared__ unsigned short ldsS[BTG * KT];
  __shared__ float yacc[BTG * 16];
  __shared__ float rsum[16];

  const int tid    = threadIdx.x;
  const int wave   = tid >> 5;
  const int lane   = tid & 31;
  const int half   = lane >> 4;
  const int r      = lane & 15;
  const int mBase  = blockIdx.x * 16;
  const int btBase = blockIdx.y * BTG;
  const bool fullM = (mBase + 16 <= N_NODES);

  if (tid < BTG * 16) yacc[tid] = 0.0f;
  if (tid < 16) {
    int gm = mBase + tid;
    rsum[tid] = (gm < N_NODES) ? rowsumA[gm] : 0.0f;
  }

  v8f acc[BTG] = {};

  for (int n0 = 0; n0 < NFULL; n0 += KT) {          // 41 guard-free chunks
    __syncthreads();
    if (fullM) s1_stage<true >(tid, n0, mBase, btBase, A, W, seq, ldsA, ldsWT, ldsS);
    else       s1_stage<false>(tid, n0, mBase, btBase, A, W, seq, ldsA, ldsWT, ldsS);
    __syncthreads();
    s1_compute(wave, half, r, ldsA, ldsWT, ldsS, acc);
  }
  {                                                 // guarded tail chunk (17 cols)
    __syncthreads();
    s1_stage<false>(tid, NFULL, mBase, btBase, A, W, seq, ldsA, ldsWT, ldsS);
    __syncthreads();
    s1_compute(wave, half, r, ldsA, ldsWT, ldsS, acc);
  }

  // Epilogue: + rowsumA*bias, relu, dot with weight2, reduce over f.
  const int   f      = wave * 16 + r;
  const float bias_f = (f < F_FEAT) ? bias[f] : 0.0f;
  const float w2_f   = (f < F_FEAT) ? w2[f]   : 0.0f;
#pragma unroll
  for (int bt = 0; bt < BTG; ++bt) {
#pragma unroll
    for (int i = 0; i < 8; ++i) {
      int m = i + half * 8;                         // C/D layout: VGPR i -> M=i (+8 hi)
      float v = acc[bt][i] + rsum[m] * bias_f;
      v = fmaxf(v, 0.0f) * w2_f;
      v += __shfl_xor(v, 1, 32);
      v += __shfl_xor(v, 2, 32);
      v += __shfl_xor(v, 4, 32);
      v += __shfl_xor(v, 8, 32);                    // sum across 16 f-lanes
      if (r == 0) atomicAdd(&yacc[bt * 16 + m], v); // ds_add_f32 across 9 waves
    }
  }
  __syncthreads();
  if (tid < BTG * 16) {
    int bt = tid >> 4, ml = tid & 15;
    int gm = mBase + ml;
    if (gm < N_NODES)
      Y1[(size_t)(btBase + bt) * N_NODES + gm] = yacc[tid] + b2[0];
  }
}

// ---------------------------------------------------------------------------
// Kernel 2: Y2[bt,m] = relu( sum_n A[m,n] * Y1[bt,n] ) as a WMMA GEMM.
// Block stages 16 bt-columns of Y1 in LDS (bf16); wave owns one m-tile.
// ---------------------------------------------------------------------------
__global__ __launch_bounds__(256) void gcn_stage2_kernel(
    const float* __restrict__ A,    // [N, N]
    const float* __restrict__ Y1,   // [BT, N]
    float* __restrict__ Y2)         // [BT, N]
{
  __shared__ unsigned short ldsY[16 * NPAD];        // 43 KB of 320 KB WGP LDS

  const int tid    = threadIdx.x;
  const int wave   = tid >> 5;
  const int lane   = tid & 31;
  const int half   = lane >> 4;
  const int r      = lane & 15;
  const int btBase = blockIdx.x * 16;               // BT = 24*16 exactly

  for (int idx = tid; idx < 16 * NPAD; idx += 256) {
    int c = idx / NPAD, n = idx - c * NPAD;
    float v = (n < N_NODES) ? Y1[(size_t)(btBase + c) * N_NODES + n] : 0.0f;
    ldsY[idx] = f32_to_bf16_bits(v);
  }
  __syncthreads();

  const int mt = blockIdx.y * 8 + wave;
  if (mt >= MTILES) return;
  const int mBase = mt * 16;
  const int gmc = min(mBase + r, N_NODES - 1);      // clamp: bad rows never stored
  const float* arow = A + (size_t)gmc * N_NODES;

  v8f acc = {};
  for (int n0 = 0; n0 < NFULL; n0 += KT) {          // guard-free main loop
    v16bf afrag, bfrag;
#pragma unroll
    for (int e = 0; e < 16; ++e) {
      afrag[e] = f2bf(arow[n0 + kidxA(e, half)]);   // contiguous 16-float run
      bfrag[e] = bits_to_bf16(ldsY[r * NPAD + n0 + e + (half << 4)]);
    }
    acc = __builtin_amdgcn_wmma_f32_16x16x32_bf16(false, afrag, false, bfrag,
                                                  (short)0, acc, false, false);
  }
  {                                                 // tail chunk (17 valid cols)
    v16bf afrag, bfrag;
#pragma unroll
    for (int e = 0; e < 16; ++e) {
      int gn = NFULL + kidxA(e, half);
      afrag[e] = (gn < N_NODES) ? f2bf(arow[gn]) : bits_to_bf16((unsigned short)0);
      bfrag[e] = bits_to_bf16(ldsY[r * NPAD + NFULL + e + (half << 4)]);
    }
    acc = __builtin_amdgcn_wmma_f32_16x16x32_bf16(false, afrag, false, bfrag,
                                                  (short)0, acc, false, false);
  }
#pragma unroll
  for (int i = 0; i < 8; ++i) {
    int gm = mBase + i + half * 8;
    if (gm < N_NODES)
      Y2[(size_t)(btBase + r) * N_NODES + gm] = fmaxf(acc[i], 0.0f);
  }
}

// ---------------------------------------------------------------------------
// Kernel 3: LSTM over T=24 steps (inherently sequential, tiny). One block,
// state in LDS, PyTorch gate order i,f,g,o. Emits sigmoid(h_last).
// ---------------------------------------------------------------------------
__global__ __launch_bounds__(1024) void lstm_kernel(
    const float* __restrict__ Y2,     // [B, T, N]
    const float* __restrict__ W_ih,   // [4H, N]
    const float* __restrict__ W_hh,   // [4H, H]
    const float* __restrict__ b_ih,   // [4H]
    const float* __restrict__ b_hh,   // [4H]
    const float* __restrict__ h0,     // [B, H]
    const float* __restrict__ c0,     // [B, H]
    float* __restrict__ sH)           // [B, H] = sigmoid(h_last)
{
  __shared__ float hS[B_BATCH * H_HID];
  __shared__ float cS[B_BATCH * H_HID];
  __shared__ float gS[B_BATCH * G4H];
  const int tid = threadIdx.x;

  if (tid < B_BATCH * H_HID) { hS[tid] = h0[tid]; cS[tid] = c0[tid]; }
  __syncthreads();

  for (int t = 0; t < T_STEPS; ++t) {
    if (tid < B_BATCH * G4H) {
      int b = tid / G4H, j = tid - b * G4H;
      const float* x   = Y2 + ((size_t)b * T_STEPS + t) * N_NODES;
      const float* wr  = W_ih + (size_t)j * N_NODES;
      float g = b_ih[j] + b_hh[j];
      for (int n = 0; n < N_NODES; ++n) g = fmaf(x[n], wr[n], g);
      const float* hr  = hS + b * H_HID;
      const float* whr = W_hh + j * H_HID;
#pragma unroll
      for (int h = 0; h < H_HID; ++h) g = fmaf(hr[h], whr[h], g);
      gS[tid] = g;
    }
    __syncthreads();
    if (tid < B_BATCH * H_HID) {
      int b = tid / H_HID, h = tid - b * H_HID;
      const float* gb = gS + b * G4H;
      float ig = gb[h], fg = gb[H_HID + h], gg = gb[2 * H_HID + h], og = gb[3 * H_HID + h];
      float c = sigf(fg) * cS[tid] + sigf(ig) * tanhf(gg);
      cS[tid] = c;
      hS[tid] = sigf(og) * tanhf(c);
    }
    __syncthreads();
  }
  if (tid < B_BATCH * H_HID) sH[tid] = sigf(hS[tid]);
}

// ---------------------------------------------------------------------------
// Kernel 4: heads  out[k,b,n] = tanh( fc_w[k,n,:] . s[b,:] + fc_b[k,n] )
// ---------------------------------------------------------------------------
__global__ void heads_kernel(const float* __restrict__ fc_w,   // [K, N, H]
                             const float* __restrict__ fc_b,   // [K, N]
                             const float* __restrict__ sH,     // [B, H]
                             float* __restrict__ out, int total)
{
  int idx = blockIdx.x * blockDim.x + threadIdx.x;
  if (idx >= total) return;
  int k   = idx / (B_BATCH * N_NODES);
  int rem = idx - k * (B_BATCH * N_NODES);
  int b   = rem / N_NODES;
  int n   = rem - b * N_NODES;
  const float* w = fc_w + ((size_t)k * N_NODES + n) * H_HID;
  const float* s = sH + b * H_HID;
  float acc = fc_b[(size_t)k * N_NODES + n];
#pragma unroll
  for (int h = 0; h < H_HID; ++h) acc = fmaf(w[h], s[h], acc);
  out[idx] = tanhf(acc);
}

// ---------------------------------------------------------------------------
extern "C" void kernel_launch(void* const* d_in, const int* in_sizes, int n_in,
                              void* d_out, int out_size, void* d_ws, size_t ws_size,
                              hipStream_t stream) {
  (void)in_sizes; (void)n_in; (void)out_size; (void)ws_size;

  const float* seq    = (const float*)d_in[0];
  const float* A      = (const float*)d_in[1];
  const float* weight = (const float*)d_in[2];
  const float* bias   = (const float*)d_in[3];
  const float* w2     = (const float*)d_in[4];
  const float* b2     = (const float*)d_in[5];
  const float* W_ih   = (const float*)d_in[6];
  const float* W_hh   = (const float*)d_in[7];
  const float* b_ih   = (const float*)d_in[8];
  const float* b_hh   = (const float*)d_in[9];
  const float* h0     = (const float*)d_in[10];
  const float* c0     = (const float*)d_in[11];
  const float* fc_w   = (const float*)d_in[12];
  const float* fc_b   = (const float*)d_in[13];
  float* out = (float*)d_out;

  // workspace layout (bytes): rowsumA | Y1 [BT,N] | Y2 [BT,N] | sH [B,H]
  char* ws = (char*)d_ws;
  float* rs = (float*)(ws);
  float* Y1 = (float*)(ws + 8192);
  float* Y2 = (float*)(ws + 8192 + 2097152);
  float* sH = (float*)(ws + 8192 + 2 * 2097152);

  rowsum_kernel<<<(N_NODES + 255) / 256, 256, 0, stream>>>(A, rs);

  dim3 g1(MTILES, BT / BTG);                     // 84 x 24 workgroups
  gcn_stage1_kernel<<<g1, 288, 0, stream>>>(seq, A, weight, bias, w2, b2, rs, Y1);

  dim3 g2(BT / 16, (MTILES + 7) / 8);            // 24 x 11 workgroups
  gcn_stage2_kernel<<<g2, 256, 0, stream>>>(A, Y1, Y2);

  lstm_kernel<<<1, 1024, 0, stream>>>(Y2, W_ih, W_hh, b_ih, b_hh, h0, c0, sH);

  int total = K_HEADS * B_BATCH * N_NODES;
  heads_kernel<<<(total + 255) / 256, 256, 0, stream>>>(fc_w, fc_b, sH, out, total);
}
